// GraphBackboneUpdate_40819369181396
// MI455X (gfx1250) — matrix operand order repaired and checked by reading, not compiled, not run on hardware
//
#include <hip/hip_runtime.h>

// ---------------------------------------------------------------------------
// GraphBackboneUpdate for MI455X (gfx1250, wave32, WMMA)
// B=8, N=1024, K=32, DIM_EDGES=256, DIM_NODES=512
// Memory-bound: 268MB edge_h streamed once through v_wmma_f32_16x16x32_bf16.
// Weights pre-transposed to bf16 column-major so B fragments load as b128s.
// A-fragment conversion is interleaved with the WMMAs (round-2 structure):
// the compiler emits clean paired v_cvt_pk_bf16_f32, and the 4 co-exec
// hazard slots hide under in-flight global_load_b128 latency.
// ---------------------------------------------------------------------------

#define BB   8
#define NN   1024
#define KK   32
#define BN   (BB * NN)          // 8192 nodes
#define BNK  (BN * KK)          // 262144 edges
#define DE   256
#define DN   512
#define EPSF 0.001f
#define SCALEF 10.0f
#define L_CO 1.2297f
#define ANG_CA_C_O 2.13847351f  // 122.52 deg in radians

typedef __bf16 v16bf __attribute__((ext_vector_type(16)));
typedef float  v8f   __attribute__((ext_vector_type(8)));
typedef float  f4    __attribute__((ext_vector_type(4)));

// ------------------------------ small math ---------------------------------

__device__ __forceinline__ void sub3(const float* a, const float* b, float* o) {
    o[0] = a[0] - b[0]; o[1] = a[1] - b[1]; o[2] = a[2] - b[2];
}
__device__ __forceinline__ float dot3(const float* a, const float* b) {
    return a[0]*b[0] + a[1]*b[1] + a[2]*b[2];
}
__device__ __forceinline__ void cross3(const float* a, const float* b, float* o) {
    o[0] = a[1]*b[2] - a[2]*b[1];
    o[1] = a[2]*b[0] - a[0]*b[2];
    o[2] = a[0]*b[1] - a[1]*b[0];
}
__device__ __forceinline__ void normed3(const float* v, float* o) {
    float inv = rsqrtf(dot3(v, v) + EPSF);
    o[0] = v[0]*inv; o[1] = v[1]*inv; o[2] = v[2]*inv;
}
__device__ __forceinline__ float sgnf(float a) { return (a >= 0.0f) ? 1.0f : -1.0f; }

// R stored row-major R[3*x+y]; columns are (u1, n2, n3)
__device__ __forceinline__ void frames3(const float* XN, const float* XCA, const float* XC,
                                        float* R) {
    float d1[3], dc[3], u1[3], uc[3], t[3], n2[3], n3[3];
    sub3(XN, XCA, d1); normed3(d1, u1);
    sub3(XC, XCA, dc); normed3(dc, uc);
    float pr = dot3(uc, u1);
    t[0] = uc[0] - pr*u1[0]; t[1] = uc[1] - pr*u1[1]; t[2] = uc[2] - pr*u1[2];
    normed3(t, n2);
    cross3(u1, n2, n3);
    #pragma unroll
    for (int x = 0; x < 3; ++x) { R[3*x+0] = u1[x]; R[3*x+1] = n2[x]; R[3*x+2] = n3[x]; }
}

__device__ __forceinline__ void rot_from_quat(const float* q, float* R) {
    float inv = rsqrtf(q[0]*q[0] + q[1]*q[1] + q[2]*q[2] + q[3]*q[3] + EPSF);
    float w = q[0]*inv, x = q[1]*inv, y = q[2]*inv, z = q[3]*inv;
    R[0] = 1.0f - 2.0f*(y*y + z*z); R[1] = 2.0f*(x*y - w*z); R[2] = 2.0f*(x*z + w*y);
    R[3] = 2.0f*(x*y + w*z); R[4] = 1.0f - 2.0f*(x*x + z*z); R[5] = 2.0f*(y*z - w*x);
    R[6] = 2.0f*(x*z - w*y); R[7] = 2.0f*(y*z + w*x); R[8] = 1.0f - 2.0f*(x*x + y*y);
}

__device__ __forceinline__ void quat_from_rot(const float* R, float* q) {
    float R00 = R[0], R11 = R[4], R22 = R[8];
    q[0] = 0.5f * sqrtf(fmaxf(1.0f + R00 + R11 + R22, 0.0f) + EPSF);
    q[1] = 0.5f * sqrtf(fmaxf(1.0f + R00 - R11 - R22, 0.0f) + EPSF) * sgnf(R[7] - R[5]);
    q[2] = 0.5f * sqrtf(fmaxf(1.0f - R00 + R11 - R22, 0.0f) + EPSF) * sgnf(R[2] - R[6]);
    q[3] = 0.5f * sqrtf(fmaxf(1.0f - R00 - R11 + R22, 0.0f) + EPSF) * sgnf(R[3] - R[1]);
}

// ------------------------------ kernels ------------------------------------

// Pack [Wq|Wt|Ww] -> WeT (16 x 256, bf16, column-major: WeT[n*256+k])
// and Wtl -> WnT (16 x 512 bf16), plus 16-wide fp32 biases.
__global__ void pack_weights(const float* __restrict__ Wq, const float* __restrict__ bq,
                             const float* __restrict__ Wt, const float* __restrict__ bt,
                             const float* __restrict__ Ww, const float* __restrict__ bw,
                             const float* __restrict__ Wtl, const float* __restrict__ btl,
                             __bf16* __restrict__ WeT, float* __restrict__ be,
                             __bf16* __restrict__ WnT, float* __restrict__ bn) {
    int t = blockIdx.x * blockDim.x + threadIdx.x;
    if (t < DE * 16) {
        int n = t / DE, k = t % DE;             // transposed layout
        float v = 0.0f;
        if (n < 4)       v = Wq[k*4 + n];
        else if (n < 7)  v = Wt[k*3 + (n-4)];
        else if (n == 7) v = Ww[k];
        WeT[t] = (__bf16)v;
    }
    if (t < DN * 16) {
        int n = t / DN, k = t % DN;
        WnT[t] = (__bf16)((n < 12) ? Wtl[k*12 + n] : 0.0f);
    }
    if (t < 16) {
        float v = 0.0f;
        if (t < 4)       v = bq[t];
        else if (t < 7)  v = bt[t-4];
        else if (t == 7) v = bw[0];
        be[t] = v;
        bn[t] = (t < 12) ? btl[t] : 0.0f;
    }
}

// Tall-skinny GEMM: out(M x OC) = A(M x KDIM) @ W(KDIM x 16) + bias.
// One wave handles TM 16-row tiles; K-step outer so the B fragment (one
// v16bf = 2 x b128 from the bf16-transposed weights) is fetched once per
// K-step and reused across TM WMMAs. Columns [SLO,SHI] scaled by `scale`.
template <int KDIM, int TM, int OC, int SLO, int SHI>
__global__ __launch_bounds__(32) void gemm_wmma(
    const float* __restrict__ A,
    const __bf16* __restrict__ WT,     // (16 x KDIM) column-major bf16
    const float* __restrict__ bias,
    float* __restrict__ out, float scale) {
    const int lane = threadIdx.x;       // 0..31
    const int hf   = lane >> 4;         // half-wave select
    const int m15  = lane & 15;
    const long row0 = (long)blockIdx.x * (16 * TM);
    const float* a0p = A + (row0 + m15) * (long)KDIM;

    v8f acc[TM];
    #pragma unroll
    for (int t = 0; t < TM; ++t) acc[t] = (v8f){};

    const __bf16* wcol = WT + (size_t)m15 * KDIM + 16 * hf;

    for (int kk = 0; kk < KDIM; kk += 32) {
        // B fragment (32x16): lane m15 = column; half 0 rows kk..kk+15,
        // half 1 rows kk+16..kk+31. 32B aligned -> 2 x global_load_b128.
        v16bf bfrag = *(const v16bf*)(wcol + kk);

        const int k0 = kk + 16 * hf;    // A half-lane K split: contiguous 16 per half (matches B)
        #pragma unroll
        for (int t = 0; t < TM; ++t) {
            const float* arow = a0p + (long)t * 16 * KDIM;
            f4 a0 = *(const f4*)(arow + k0);
            f4 a1 = *(const f4*)(arow + k0 + 4);
            f4 a2 = *(const f4*)(arow + k0 + 8);
            f4 a3 = *(const f4*)(arow + k0 + 12);
            v16bf afrag;
            #pragma unroll
            for (int i = 0; i < 4; ++i) {
                afrag[i]      = (__bf16)a0[i];
                afrag[4 + i]  = (__bf16)a1[i];
                afrag[8 + i]  = (__bf16)a2[i];
                afrag[12 + i] = (__bf16)a3[i];
            }
            acc[t] = __builtin_amdgcn_wmma_f32_16x16x32_bf16(
                false, afrag, false, bfrag, (short)0, acc[t], false, false);
        }
    }

    // D layout: VGPR r, lanes 0-15 -> M=r, lanes 16-31 -> M=r+8; N = lane&15.
    if (m15 < OC) {
        const float bval = bias[m15];
        const float s = (m15 >= SLO && m15 <= SHI) ? scale : 1.0f;
        #pragma unroll
        for (int t = 0; t < TM; ++t) {
            float* orow = out + (row0 + t * 16 + 8 * hf) * OC + m15;
            #pragma unroll
            for (int r = 0; r < 8; ++r)
                orow[(long)r * OC] = s * (acc[t][r] + bval);
        }
    }
}

// Initial backbone frames, masked by C>0.
__global__ void frames_init(const float* __restrict__ X, const int* __restrict__ C,
                            float* __restrict__ fR, float* __restrict__ ft) {
    int idx = blockIdx.x * blockDim.x + threadIdx.x;
    if (idx >= BN) return;
    const float* x = X + (size_t)idx * 12;       // atoms N, CA, C, O
    float R[9];
    frames3(x + 0, x + 3, x + 6, R);
    float mask = (C[idx] > 0) ? 1.0f : 0.0f;
    #pragma unroll
    for (int i = 0; i < 9; ++i) fR[(size_t)idx*9 + i] = mask * R[i];
    #pragma unroll
    for (int i = 0; i < 3; ++i) ft[(size_t)idx*3 + i] = mask * x[3 + i];
}

// Softmax-weighted transform averaging; top eigenvector of the 4x4 quat
// accumulator via power iteration (A is PSD: sum of p * q q^T).
__global__ void avg_transforms(const float* __restrict__ eo,     // (BNK, 8) q|t|logit
                               const float* __restrict__ fR, const float* __restrict__ ft,
                               const float* __restrict__ mask_ij,
                               const int* __restrict__ edge_idx,
                               float* __restrict__ oR, float* __restrict__ ot) {
    int idx = blockIdx.x * blockDim.x + threadIdx.x;
    if (idx >= BN) return;
    int b = idx / NN;

    // masked softmax over K edges
    float lg[KK];
    float mx = -1.0e9f;
    for (int k = 0; k < KK; ++k) {
        size_t e = (size_t)idx * KK + k;
        float l = (mask_ij[e] > 0.0f) ? eo[e*8 + 7] : -1.0e9f;
        lg[k] = l;
        mx = fmaxf(mx, l);
    }
    float sum = 0.0f;
    for (int k = 0; k < KK; ++k) { lg[k] = expf(lg[k] - mx); sum += lg[k]; }
    float inv_sum = 1.0f / sum;

    float Aq[16];
    #pragma unroll
    for (int i = 0; i < 16; ++i) Aq[i] = 0.0f;
    float tav[3] = {0.0f, 0.0f, 0.0f};

    for (int k = 0; k < KK; ++k) {
        size_t e = (size_t)idx * KK + k;
        float p = lg[k] * inv_sum;
        int j = edge_idx[e];
        const float* Rj = fR + (size_t)(b * NN + j) * 9;
        const float* tj = ft + (size_t)(b * NN + j) * 3;
        const float* ev = eo + e * 8;

        float Rji[9];
        rot_from_quat(ev, Rji);                  // ev[0..3] = predicted quat
        float Rp[9];
        #pragma unroll
        for (int x = 0; x < 3; ++x)
            #pragma unroll
            for (int y = 0; y < 3; ++y)
                Rp[3*x+y] = Rj[3*x+0]*Rji[0+y] + Rj[3*x+1]*Rji[3+y] + Rj[3*x+2]*Rji[6+y];
        float tp[3];
        #pragma unroll
        for (int x = 0; x < 3; ++x)
            tp[x] = Rj[3*x+0]*ev[4] + Rj[3*x+1]*ev[5] + Rj[3*x+2]*ev[6] + tj[x];

        float qk[4];
        quat_from_rot(Rp, qk);
        #pragma unroll
        for (int x = 0; x < 3; ++x) tav[x] += p * tp[x];
        #pragma unroll
        for (int i = 0; i < 4; ++i)
            #pragma unroll
            for (int j2 = 0; j2 < 4; ++j2)
                Aq[4*i + j2] += p * qk[i] * qk[j2];
    }

    // power iteration: start from column with the largest diagonal entry
    int im = 0;
    for (int i = 1; i < 4; ++i) if (Aq[5*i] > Aq[5*im]) im = i;
    float v[4];
    #pragma unroll
    for (int i = 0; i < 4; ++i) v[i] = Aq[4*i + im];
    for (int it = 0; it < 32; ++it) {
        float w[4];
        #pragma unroll
        for (int i = 0; i < 4; ++i)
            w[i] = Aq[4*i+0]*v[0] + Aq[4*i+1]*v[1] + Aq[4*i+2]*v[2] + Aq[4*i+3]*v[3];
        float inv = rsqrtf(w[0]*w[0] + w[1]*w[1] + w[2]*w[2] + w[3]*w[3] + 1.0e-20f);
        #pragma unroll
        for (int i = 0; i < 4; ++i) v[i] = w[i] * inv;
    }

    float Rn[9];
    rot_from_quat(v, Rn);
    #pragma unroll
    for (int i = 0; i < 9; ++i) oR[(size_t)idx*9 + i] = Rn[i];
    #pragma unroll
    for (int i = 0; i < 3; ++i) ot[(size_t)idx*3 + i] = tav[i];
}

// X_chain[a][x] = sum_y R[x][y] * T_ATOM[a][y] + t[x]  for atoms N, CA, C.
__global__ void build_chain(const float* __restrict__ oR, const float* __restrict__ ot,
                            float* __restrict__ chain) {
    int idx = blockIdx.x * blockDim.x + threadIdx.x;
    if (idx >= BN) return;
    const float* R = oR + (size_t)idx * 9;
    const float* t = ot + (size_t)idx * 3;
    const float TA[3][3] = {{1.459f, 0.0f, 0.0f},
                            {0.0f,   0.0f, 0.0f},
                            {-0.547f, 0.0f, -1.424f}};
    float* ch = chain + (size_t)idx * 9;
    #pragma unroll
    for (int a = 0; a < 3; ++a)
        #pragma unroll
        for (int x = 0; x < 3; ++x)
            ch[a*3 + x] = R[3*x+0]*TA[a][0] + R[3*x+1]*TA[a][1] + R[3*x+2]*TA[a][2] + t[x];
}

// O-atom extension (phi = pi), masking, second frames, local t update.
__global__ void finalize(const float* __restrict__ chain,
                         const float* __restrict__ node_out,  // (BN, 12)
                         const int* __restrict__ C,
                         float* __restrict__ out) {
    int idx = blockIdx.x * blockDim.x + threadIdx.x;
    if (idx >= BN) return;
    int n = idx % NN;
    const float* ch = chain + (size_t)idx * 9;
    float XN[3]  = {ch[0], ch[1], ch[2]};
    float XCA[3] = {ch[3], ch[4], ch[5]};
    float XC[3]  = {ch[6], ch[7], ch[8]};

    float XNn[3] = {0.0f, 0.0f, 0.0f};
    if (n < NN - 1 && C[idx + 1] > 0) {
        const float* cn = chain + (size_t)(idx + 1) * 9;
        XNn[0] = cn[0]; XNn[1] = cn[1]; XNn[2] = cn[2];
    }

    // extend_atom(Xa=XNn, Xb=XCA, Xc=XC); cos(pi)=-1, sin(pi)=0
    float dcb[3], bc[3], ab[3], cr[3], nv[3], m[3];
    sub3(XC, XCA, dcb); normed3(dcb, bc);
    sub3(XCA, XNn, ab);
    cross3(ab, bc, cr); normed3(cr, nv);
    cross3(nv, bc, m);
    const float ct = cosf(ANG_CA_C_O), st = sinf(ANG_CA_C_O);
    float XO[3];
    #pragma unroll
    for (int x = 0; x < 3; ++x)
        XO[x] = XC[x] + L_CO * (-ct * bc[x] - st * m[x]);

    float mask = (C[idx] > 0) ? 1.0f : 0.0f;
    float X4[4][3];
    #pragma unroll
    for (int x = 0; x < 3; ++x) {
        X4[0][x] = mask * XN[x];
        X4[1][x] = mask * XCA[x];
        X4[2][x] = mask * XC[x];
        X4[3][x] = mask * XO[x];
    }

    float R2[9];
    frames3(X4[0], X4[1], X4[2], R2);

    const float* tl = node_out + (size_t)idx * 12;
    float* o = out + (size_t)idx * 12;
    #pragma unroll
    for (int a = 0; a < 4; ++a)
        #pragma unroll
        for (int x = 0; x < 3; ++x)
            o[a*3 + x] = X4[a][x] +
                mask * (R2[3*x+0]*tl[a*3+0] + R2[3*x+1]*tl[a*3+1] + R2[3*x+2]*tl[a*3+2]);
}

// ------------------------------ launcher -----------------------------------

extern "C" void kernel_launch(void* const* d_in, const int* in_sizes, int n_in,
                              void* d_out, int out_size, void* d_ws, size_t ws_size,
                              hipStream_t stream) {
    const float* X        = (const float*)d_in[0];
    const float* node_h   = (const float*)d_in[1];
    const float* edge_h   = (const float*)d_in[2];
    // d_in[3] = mask_i (unused by the reference math)
    const float* mask_ij  = (const float*)d_in[4];
    const float* Wq  = (const float*)d_in[5];
    const float* bq  = (const float*)d_in[6];
    const float* Wt  = (const float*)d_in[7];
    const float* bt  = (const float*)d_in[8];
    const float* Ww  = (const float*)d_in[9];
    const float* bw  = (const float*)d_in[10];
    const float* Wtl = (const float*)d_in[11];
    const float* btl = (const float*)d_in[12];
    const int* C        = (const int*)d_in[13];
    const int* edge_idx = (const int*)d_in[14];
    float* out = (float*)d_out;

    char* ws = (char*)d_ws;
    size_t off = 0;
    __bf16* WeT     = (__bf16*)(ws + off); off += (size_t)DE * 16 * 2;  // 8 KB
    __bf16* WnT     = (__bf16*)(ws + off); off += (size_t)DN * 16 * 2;  // 16 KB
    float* be       = (float*)(ws + off); off += 16 * 4;
    float* bn       = (float*)(ws + off); off += 16 * 4;
    float* edge_out = (float*)(ws + off); off += (size_t)BNK * 8 * 4;   // 8 MB
    float* fR       = (float*)(ws + off); off += (size_t)BN * 9 * 4;
    float* ft       = (float*)(ws + off); off += (size_t)BN * 3 * 4;
    float* oR       = (float*)(ws + off); off += (size_t)BN * 9 * 4;
    float* ot       = (float*)(ws + off); off += (size_t)BN * 3 * 4;
    float* node_out = (float*)(ws + off); off += (size_t)BN * 12 * 4;
    float* chain    = (float*)(ws + off); off += (size_t)BN * 9 * 4;

    pack_weights<<<32, 256, 0, stream>>>(Wq, bq, Wt, bt, Ww, bw, Wtl, btl,
                                         WeT, be, WnT, bn);

    // Fused edge projection: (BNK x 256) @ (256 x 16) -> q(0-3), SCALE*t(4-6), logit(7)
    // TM=4 -> 4096 waves streaming 64KB each.
    gemm_wmma<DE, 4, 8, 4, 6><<<BNK / (16 * 4), 32, 0, stream>>>(
        edge_h, WeT, be, edge_out, SCALEF);

    // Node projection: (BN x 512) @ (512 x 16) -> t_local(0-11). TM=2 -> 256 waves.
    gemm_wmma<DN, 2, 12, 16, -1><<<BN / (16 * 2), 32, 0, stream>>>(
        node_h, WnT, bn, node_out, 1.0f);

    frames_init<<<BN / 256, 256, 0, stream>>>(X, C, fR, ft);

    avg_transforms<<<BN / 256, 256, 0, stream>>>(edge_out, fR, ft, mask_ij,
                                                 edge_idx, oR, ot);

    build_chain<<<BN / 256, 256, 0, stream>>>(oR, ot, chain);

    finalize<<<BN / 256, 256, 0, stream>>>(chain, node_out, C, out);
}
